// S4Model_39831526703858
// MI455X (gfx1250) — matrix operand order repaired and checked by reading, not compiled, not run on hardware
//
#include <hip/hip_runtime.h>

// Problem constants (match reference)
constexpr int Bq   = 16;
constexpr int Lq   = 2048;
constexpr int Hq   = 512;
constexpr int N2q  = 32;
constexpr int OUTq = 256;
constexpr int NLq  = 4;

typedef __attribute__((ext_vector_type(16))) __bf16 v16bf;
typedef __attribute__((ext_vector_type(8)))  __bf16 v8bf;
typedef __attribute__((ext_vector_type(8)))  float  v8f;

typedef unsigned int u32x4 __attribute__((ext_vector_type(4)));
typedef int          i32x4 __attribute__((ext_vector_type(4)));
typedef int          i32x8 __attribute__((ext_vector_type(8)));

// Tensor Data Mover availability (device pass only; host pass lacks amdgcn
// builtins so __has_builtin is false there and the manual path compiles).
#if defined(__gfx1250__) && __has_builtin(__builtin_amdgcn_tensor_load_to_lds) \
    && __has_builtin(__builtin_amdgcn_s_wait_tensorcnt)
#define USE_TDM 1
#else
#define USE_TDM 0
#endif

// ---------------------------------------------------------------------------
// One-time weight conversion f32 -> bf16 (weights then live in L2: 2.25 MB)
// ---------------------------------------------------------------------------
__global__ __launch_bounds__(256) void cvt_bf16_kernel(
    const float* __restrict__ src, __bf16* __restrict__ dst, int n) {
  int i = blockIdx.x * 256 + threadIdx.x;
  if (i < n) dst[i] = (__bf16)src[i];
}

// ---------------------------------------------------------------------------
// Embedding gather: writes f32 (residual path) + bf16 (GEMM A path)
// ---------------------------------------------------------------------------
__global__ __launch_bounds__(256) void embed_kernel(
    const int* __restrict__ x, const float* __restrict__ emb,
    float* __restrict__ h, __bf16* __restrict__ hb) {
  size_t idx = (size_t)blockIdx.x * 256 + threadIdx.x;  // over B*L*H
  int tok = x[idx >> 9];                                // H = 512 = 2^9
  float v = emb[((size_t)tok << 9) + (idx & 511u)];
  h[idx] = v;
  hb[idx] = (__bf16)v;
}

#if USE_TDM
// Issue a TDM copy of a 128x64 bf16 tile (row stride 512 elems in global)
// into LDS at lds_off with a padded 72-element row stride (pad fields:
// 32 DWORDs per row, then 4 DWORDs pad).
__device__ __forceinline__ void tdm_issue_a(const __bf16* gsrc,
                                            unsigned lds_off) {
  const unsigned long long gaddr = (unsigned long long)gsrc;
  u32x4 g0;
  g0[0] = 1u;                                   // count=1, user descriptor
  g0[1] = lds_off;                              // lds_addr (byte offset)
  g0[2] = (unsigned)gaddr;                      // global_addr[31:0]
  g0[3] = (unsigned)(gaddr >> 32) | (2u << 30); // global_addr[56:32] | type=2
  i32x8 g1;
  // data_size=1(2B) | pad_enable | pad_interval=4(32 DW) | pad_amount=3(4 DW)
  g1[0] = (int)((1u << 16) | (1u << 20) | (4u << 22) | (3u << 25));
  g1[1] = (int)(512u << 16);                    // tensor_dim0 = 512
  g1[2] = (int)(32768u << 16);                  // tensor_dim1 = 32768
  g1[3] = (int)(64u << 16);                     // tile_dim0 = 64
  g1[4] = (int)128;                             // tile_dim1 = 128
  g1[5] = (int)512;                             // tensor_dim0_stride = 512
  g1[6] = 0;
  g1[7] = 0;
  i32x4 z4 = {0, 0, 0, 0};
#if __clang_major__ >= 23
  i32x8 z8 = {0, 0, 0, 0, 0, 0, 0, 0};
  __builtin_amdgcn_tensor_load_to_lds(g0, g1, z4, z4, z8, 0);
#else
  __builtin_amdgcn_tensor_load_to_lds(g0, g1, z4, z4, 0);
#endif
}
#endif

// ---------------------------------------------------------------------------
// Fused GEMM:  out = f( A[M,512] @ W[512,N] + bias , resid )
//   mode 0: y+=resid; y = relu(y)+y; y = 2*y   (layer linear + res chain)
//   mode 1: y only                              (final projection)
// A and W already bf16. Block: 256 thr = 8 waves; block tile 128(M) x 64(N);
// wave tile 32x32 (4 accumulators); K slab 64 -> 8 WMMAs per slab.
// Double-buffered async pipeline: wave 0 issues the Tensor Data Mover copy of
// the NEXT A slab, all threads stage the NEXT W slab, compute runs on the
// current buffers, then the issuing wave drains TENSORcnt and one barrier
// flips the buffers (DMA + staging fully overlap the WMMAs).
// ---------------------------------------------------------------------------
__global__ __launch_bounds__(256) void gemm_bf16_kernel(
    const __bf16* __restrict__ Ab, const __bf16* __restrict__ Wb,
    const float* __restrict__ bias, const float* __restrict__ resid,
    float* __restrict__ out, int Ntot, int mode) {
  __shared__ __align__(16) __bf16 ldsA[2][128][72];  // [m][k], pad 64->72
  __shared__ __align__(16) __bf16 ldsB[2][64][72];   // transposed: [n][k]

  const int tid  = threadIdx.x;
  const int lane = tid & 31;
  const int wid  = tid >> 5;
  const int wm   = wid & 3;        // wave M index (0..3)
  const int wn   = wid >> 2;       // wave N index (0..1)
  const int hi   = lane >> 4;      // lane half (ISA fragment layouts)
  const int lo   = lane & 15;

  const int m_blk = blockIdx.y * 128;
  const int n_blk = blockIdx.x * 64;
  const int m0 = wm * 32;
  const int n0 = wn * 32;

  // staging thread mapping
  const int ar = tid >> 1;           // 0..127 (A row)       (manual path)
  const int ac = (tid & 1) << 5;     // 0 or 32 (A col base)  (manual path)
  const int br = tid & 63;           // 0..63  (W row = k)
  const int bc = (tid >> 6) << 4;    // 0,16,32,48 (W col base, 16 bf16)

  v8f acc[2][2] = {};

  // ---- prologue: fill buffer 0 with slab k0 = 0 ----
  {
    const __bf16* gw = Wb + (size_t)br * Ntot + n_blk + bc;
    const v8bf w0 = *(const v8bf*)gw;
    const v8bf w1 = *(const v8bf*)(gw + 8);
#pragma unroll
    for (int j = 0; j < 8; ++j) {
      ldsB[0][bc + j][br]     = w0[j];
      ldsB[0][bc + 8 + j][br] = w1[j];
    }
#if USE_TDM
    if (wid == 0) {
      tdm_issue_a(Ab + (size_t)m_blk * 512,
                  (unsigned)(unsigned long long)&ldsA[0][0][0]);
      __builtin_amdgcn_s_wait_tensorcnt((short)0);
    }
#else
    const __bf16* ga = Ab + (size_t)(m_blk + ar) * 512 + ac;
#pragma unroll
    for (int c = 0; c < 4; ++c)
      *(v8bf*)&ldsA[0][ar][ac + c * 8] = *(const v8bf*)(ga + c * 8);
#endif
  }
  __syncthreads();

  for (int k0 = 0; k0 < 512; k0 += 64) {
    const int cur = (k0 >> 6) & 1;
    const int nxt = cur ^ 1;
    const bool more = (k0 + 64) < 512;

    // ---- kick off next slab into the back buffer (overlaps compute) ----
    if (more) {
#if USE_TDM
      if (wid == 0)
        tdm_issue_a(Ab + (size_t)m_blk * 512 + k0 + 64,
                    (unsigned)(unsigned long long)&ldsA[nxt][0][0]);
#else
      const __bf16* ga = Ab + (size_t)(m_blk + ar) * 512 + k0 + 64 + ac;
#pragma unroll
      for (int c = 0; c < 4; ++c)
        *(v8bf*)&ldsA[nxt][ar][ac + c * 8] = *(const v8bf*)(ga + c * 8);
#endif
      const __bf16* gw = Wb + (size_t)(k0 + 64 + br) * Ntot + n_blk + bc;
      const v8bf w0 = *(const v8bf*)gw;
      const v8bf w1 = *(const v8bf*)(gw + 8);
#pragma unroll
      for (int j = 0; j < 8; ++j) {
        ldsB[nxt][bc + j][br]     = w0[j];
        ldsB[nxt][bc + 8 + j][br] = w1[j];
      }
      if (k0 + 128 < 512)          // prefetch slab after next (global_prefetch)
        __builtin_prefetch(gw + (size_t)64 * Ntot, 0, 0);
    }

    // ---- compute on the front buffer: 2 K-steps of 32, 8 WMMAs ----
    // A elem j: K = 2*(j&8) + 8*hi + (j&7)  -> two contiguous 8-elt chunks
    // B elem j: K = 2*(j&8) + 8*hi + (j&7)  -> same as A
#pragma unroll
    for (int kk = 0; kk < 64; kk += 32) {
      v16bf afr[2], bfr[2];
#pragma unroll
      for (int t = 0; t < 2; ++t) {
        const v8bf a0 = *(const v8bf*)&ldsA[cur][m0 + t * 16 + lo][kk + hi * 8];
        const v8bf a1 =
            *(const v8bf*)&ldsA[cur][m0 + t * 16 + lo][kk + 16 + hi * 8];
        const v8bf b0 = *(const v8bf*)&ldsB[cur][n0 + t * 16 + lo][kk + hi * 8];
        const v8bf b1 =
            *(const v8bf*)&ldsB[cur][n0 + t * 16 + lo][kk + 16 + hi * 8];
#pragma unroll
        for (int j = 0; j < 8; ++j) {
          afr[t][j] = a0[j]; afr[t][j + 8] = a1[j];
          bfr[t][j] = b0[j]; bfr[t][j + 8] = b1[j];
        }
      }
#pragma unroll
      for (int mi = 0; mi < 2; ++mi)
#pragma unroll
        for (int ni = 0; ni < 2; ++ni)
          acc[mi][ni] = __builtin_amdgcn_wmma_f32_16x16x32_bf16(
              false, afr[mi], false, bfr[ni], (short)0, acc[mi][ni],
              false, false);
    }

#if USE_TDM
    if (more && wid == 0)          // only the issuing wave has TENSORcnt > 0
      __builtin_amdgcn_s_wait_tensorcnt((short)0);
#endif
    __syncthreads();               // flip buffers
  }

  // --- epilogue: D elem r -> (m = 8*hi + r, n = lo) within each 16x16 ---
#pragma unroll
  for (int mi = 0; mi < 2; ++mi) {
#pragma unroll
    for (int ni = 0; ni < 2; ++ni) {
#pragma unroll
      for (int r = 0; r < 8; ++r) {
        const int gm = m_blk + m0 + mi * 16 + hi * 8 + r;
        const int gn = n_blk + n0 + ni * 16 + lo;
        float y = acc[mi][ni][r] + bias[gn];
        if (mode == 0) {
          const float hin = resid[(size_t)gm * 512 + gn];
          const float h1 = y + hin;
          const float h2 = fmaxf(h1, 0.f) + h1;   // relu + residual
          out[(size_t)gm * Ntot + gn] = 2.f * h2; // identity + residual
        } else {
          out[(size_t)gm * Ntot + gn] = y;
        }
      }
    }
  }
}

// ---------------------------------------------------------------------------
// S4D as a diagonal complex recurrence (exact rewrite of the FFT causal conv):
//   s_l = e^{dtA} s_{l-1} + u_l ;  y_l = 2*Re(sum_n coef_n s_{n,l}) + D u_l
// One wave per (b,h) sequence; lane n holds state n (N2 == 32 == wave32).
// Fuses +u residual, relu+residual, doubling; dual-writes f32 + bf16.
// ---------------------------------------------------------------------------
__global__ __launch_bounds__(256) void s4d_kernel(
    const float* __restrict__ u, const float* __restrict__ log_dt,
    const float* __restrict__ log_A_real, const float* __restrict__ A_imag,
    const float* __restrict__ C_re, const float* __restrict__ C_im,
    const float* __restrict__ Ds, float* __restrict__ out,
    __bf16* __restrict__ outb) {
  const int tid  = threadIdx.x;
  const int lane = tid & 31;
  const int wg   = blockIdx.x * 8 + (tid >> 5);  // 0 .. B*H-1
  const int b    = wg >> 9;                      // / 512
  const int h    = wg & 511;

  const float dt = __expf(log_dt[h]);
  const float Ar = -__expf(log_A_real[h * N2q + lane]);
  const float Ai = A_imag[h * N2q + lane];
  const float em = __expf(Ar * dt);
  const float eAr = em * __cosf(Ai * dt);        // e^{dtA}
  const float eAi = em * __sinf(Ai * dt);
  // coef = C * (e^{dtA} - 1) / A
  const float numr = eAr - 1.f, numi = eAi;
  const float den  = Ar * Ar + Ai * Ai;
  const float fr = (numr * Ar + numi * Ai) / den;
  const float fi = (numi * Ar - numr * Ai) / den;
  const float Cr = C_re[h * N2q + lane], Ci = C_im[h * N2q + lane];
  const float cr = Cr * fr - Ci * fi;
  const float ci = Cr * fi + Ci * fr;
  const float Dv = Ds[h];

  const size_t base = (size_t)b * Lq * Hq + h;
  const float* up  = u    + base;
  float*       op  = out  + base;
  __bf16*      opb = outb + base;

  float sr = 0.f, si = 0.f;
  for (int l = 0; l < Lq; ++l) {
    const float ul = up[(size_t)l * Hq];         // uniform across wave
    const float nsr = eAr * sr - eAi * si + ul;
    const float nsi = eAr * si + eAi * sr;
    sr = nsr; si = nsi;
    float p = cr * sr - ci * si;                 // Re(coef * s)
#pragma unroll
    for (int off = 16; off > 0; off >>= 1) p += __shfl_xor(p, off);
    const float y  = 2.f * p + Dv * ul;          // s4d output (incl. D skip)
    const float h4 = y + ul;                     // + residual
    const float h5 = fmaxf(h4, 0.f) + h4;        // relu + residual
    const float h6 = 2.f * h5;
    if (lane == 0) {
      op[(size_t)l * Hq]  = h6;
      opb[(size_t)l * Hq] = (__bf16)h6;
    }
  }
}

// ---------------------------------------------------------------------------
extern "C" void kernel_launch(void* const* d_in, const int* in_sizes, int n_in,
                              void* d_out, int out_size, void* d_ws,
                              size_t ws_size, hipStream_t stream) {
  const int*   x   = (const int*)  d_in[0];
  const float* emb = (const float*)d_in[1];
  const float* Ws  = (const float*)d_in[2];
  const float* bs  = (const float*)d_in[3];
  const float* ldt = (const float*)d_in[4];
  const float* lAr = (const float*)d_in[5];
  const float* Aim = (const float*)d_in[6];
  const float* Cre = (const float*)d_in[7];
  const float* Cim = (const float*)d_in[8];
  const float* Dsp = (const float*)d_in[9];
  const float* Wo  = (const float*)d_in[10];
  const float* bo  = (const float*)d_in[11];

  const size_t act = (size_t)Bq * Lq * Hq;       // 16.7M elements
  float*  hA  = (float*)d_ws;                    // f32 activations (64 MiB)
  float*  uB  = hA + act;                        // f32 u buffer     (64 MiB)
  __bf16* hAb = (__bf16*)(uB + act);             // bf16 activations (32 MiB)
  __bf16* wsb = hAb + act;                       // bf16 weights (2.25 MiB)
  __bf16* wob = wsb + (size_t)NLq * Hq * Hq;

  const int nWs = NLq * Hq * Hq;                 // 1,048,576
  const int nWo = Hq * OUTq;                     // 131,072
  cvt_bf16_kernel<<<(nWs + 255) / 256, 256, 0, stream>>>(Ws, wsb, nWs);
  cvt_bf16_kernel<<<(nWo + 255) / 256, 256, 0, stream>>>(Wo, wob, nWo);

  embed_kernel<<<(int)(act / 256), 256, 0, stream>>>(x, emb, hA, hAb);

  for (int i = 0; i < NLq; ++i) {
    gemm_bf16_kernel<<<dim3(Hq / 64, (Bq * Lq) / 128), 256, 0, stream>>>(
        hAb, wsb + (size_t)i * Hq * Hq, bs + i * Hq, hA, uB, Hq, 0);
    s4d_kernel<<<(Bq * Hq) / 8, 256, 0, stream>>>(
        uB, ldt + i * Hq, lAr + (size_t)i * Hq * N2q,
        Aim + (size_t)i * Hq * N2q, Cre + (size_t)i * Hq * N2q,
        Cim + (size_t)i * Hq * N2q, Dsp + i * Hq, hA, hAb);
  }

  gemm_bf16_kernel<<<dim3(OUTq / 64, (Bq * Lq) / 128), 256, 0, stream>>>(
      hAb, wob, bo, nullptr, (float*)d_out, OUTq, 1);
}
